// PairedKidneyBackbone_86028194938912
// MI455X (gfx1250) — matrix operand order repaired and checked by reading, not compiled, not run on hardware
//
#include <hip/hip_runtime.h>
#include <hip/hip_bf16.h>

typedef _Float16 half_t;
typedef __attribute__((ext_vector_type(16))) _Float16 v16h;
typedef __attribute__((ext_vector_type(8)))  float    v8f;

#define NEG_SLOPE 0.2f
#define LN_EPS 1e-5f

// ---------- helpers ----------
__device__ __forceinline__ float wred(float v) {
  #pragma unroll
  for (int o = 16; o > 0; o >>= 1) v += __shfl_xor(v, o, 32);
  return v;
}
// order-preserving float -> u32 map (for atomicMax on floats incl. negatives)
__device__ __forceinline__ unsigned enc_f32(float f) {
  unsigned u = __float_as_uint(f);
  return (u & 0x80000000u) ? ~u : (u | 0x80000000u);
}
__device__ __forceinline__ float dec_f32(unsigned u) {
  unsigned b = (u & 0x80000000u) ? (u & 0x7fffffffu) : ~u;
  return __uint_as_float(b);
}

// ---------- K0: fold the linear chain (runs once per call, 1 block) ----------
__global__ void prep_kernel(const float* __restrict__ w1, const float* __restrict__ b1,
                            const float* __restrict__ w2, const float* __restrict__ b2,
                            const float* __restrict__ gwl, const float* __restrict__ gwr,
                            const float* __restrict__ wf1, const float* __restrict__ wf2,
                            float* cW12, float* cbx,
                            float* cWl, float* cbl, float* cWr, float* cbr,
                            half_t* wff1h, half_t* wff2h) {
  __shared__ float sW12[192];
  __shared__ float sbx[64];
  int tid = threadIdx.x;
  if (tid < 192) {
    int i = tid >> 6, j = tid & 63;
    float s = 0.f;
    for (int k = 0; k < 64; k++) s += w1[i * 64 + k] * w2[k * 64 + j];
    sW12[tid] = s; cW12[tid] = s;
  } else if (tid < 256) {
    int j = tid - 192;
    float s = b2[j];
    for (int k = 0; k < 64; k++) s += b1[k] * w2[k * 64 + j];
    sbx[j] = s; cbx[j] = s;
  }
  __syncthreads();
  for (int idx = tid; idx < 768; idx += 256) {
    int i = idx / 256, c = idx % 256;
    float sl = 0.f, sr = 0.f;
    for (int k = 0; k < 64; k++) {
      float w = sW12[i * 64 + k];
      sl += w * gwl[k * 256 + c];
      sr += w * gwr[k * 256 + c];
    }
    cWl[idx] = sl; cWr[idx] = sr;
  }
  for (int c = tid; c < 256; c += 256) {
    float sl = 0.f, sr = 0.f;
    for (int k = 0; k < 64; k++) {
      float b = sbx[k];
      sl += b * gwl[k * 256 + c];
      sr += b * gwr[k * 256 + c];
    }
    cbl[c] = sl; cbr[c] = sr;
  }
  for (int i = tid; i < 4096; i += 256) {
    wff1h[i] = (half_t)wf1[i];
    wff2h[i] = (half_t)wf2[i];
  }
}

// ---------- K0b: re-init atomic buffers every call ----------
__global__ void init_kernel(unsigned* smax, float* denom, float* accum, int N) {
  int stride = gridDim.x * blockDim.x;
  int i = blockIdx.x * blockDim.x + threadIdx.x;
  int totA = N * 256;
  for (int idx = i; idx < totA; idx += stride) accum[idx] = 0.f;
  int totD = N * 4;
  for (int idx = i; idx < totD; idx += stride) {
    denom[idx] = 0.f;
    smax[idx] = enc_f32(-3.0e38f);
  }
}

// ---------- K1: node features + folded projections (wave per node) ----------
__global__ void node_kernel(const float* __restrict__ arr, const float* __restrict__ dep,
                            const float* __restrict__ hardm, const int* __restrict__ ts,
                            const float* __restrict__ cW12, const float* __restrict__ cbx,
                            const float* __restrict__ cWl, const float* __restrict__ cbl,
                            const float* __restrict__ cWr, const float* __restrict__ cbr,
                            float* __restrict__ x, float* __restrict__ xl, float* __restrict__ xr,
                            int N) {
  __shared__ float sW12[192], sbx[64], sWl[768], sbl[256], sWr[768], sbr[256];
  int tid = threadIdx.x;
  for (int i = tid; i < 192; i += 256) sW12[i] = cW12[i];
  for (int i = tid; i < 64;  i += 256) sbx[i]  = cbx[i];
  for (int i = tid; i < 768; i += 256) { sWl[i] = cWl[i]; sWr[i] = cWr[i]; }
  for (int i = tid; i < 256; i += 256) { sbl[i] = cbl[i]; sbr[i] = cbr[i]; }
  __syncthreads();
  int wave = tid >> 5, lane = tid & 31;
  int n = blockIdx.x * 8 + wave;
  if (n >= N) return;
  float t = (float)(*ts);
  float a = arr[n], d = dep[n], h = hardm[n];
  float td = fmaxf(d - a, 1.f);
  float p = (t - a) / td;
  float leave = ((d - t) <= 1.f) ? 1.f : 0.f;
  #pragma unroll
  for (int s = 0; s < 2; s++) {
    int j = lane + 32 * s;
    x[(size_t)n * 64 + j] = p * sW12[j] + h * sW12[64 + j] + leave * sW12[128 + j] + sbx[j];
  }
  #pragma unroll
  for (int s = 0; s < 8; s++) {
    int c = lane + 32 * s;
    xl[(size_t)n * 256 + c] = p * sWl[c] + h * sWl[256 + c] + leave * sWl[512 + c] + sbl[c];
    xr[(size_t)n * 256 + c] = p * sWr[c] + h * sWr[256 + c] + leave * sWr[512 + c] + sbr[c];
  }
}

// ---------- K2: per-edge attention scores + segment max (wave per edge) ----------
__global__ void edge_score_kernel(const int* __restrict__ ei,
                                  const float* __restrict__ xl, const float* __restrict__ xr,
                                  const float* __restrict__ att,
                                  float* __restrict__ score, unsigned* __restrict__ smax,
                                  int E, int Etot) {
  int wave = threadIdx.x >> 5, lane = threadIdx.x & 31;
  int e = blockIdx.x * 8 + wave;
  if (e >= Etot) return;
  int src, dst;
  if (e < E) { src = ei[e]; dst = ei[E + e]; }
  else       { src = dst = e - E; }  // self loops
  const float* pl = xl + (size_t)src * 256;
  const float* pr = xr + (size_t)dst * 256;
  #pragma unroll
  for (int h = 0; h < 4; h++) {
    float v0 = pl[h * 64 + lane]      + pr[h * 64 + lane];
    float v1 = pl[h * 64 + lane + 32] + pr[h * 64 + lane + 32];
    v0 = (v0 > 0.f) ? v0 : NEG_SLOPE * v0;
    v1 = (v1 > 0.f) ? v1 : NEG_SLOPE * v1;
    float s = v0 * att[h * 64 + lane] + v1 * att[h * 64 + lane + 32];
    s = wred(s);
    if (lane == 0) {
      score[(size_t)e * 4 + h] = s;
      atomicMax(&smax[dst * 4 + h], enc_f32(s));
    }
  }
}

// ---------- K3: softmax numerators/denominators via scatter atomics ----------
__global__ void edge_accum_kernel(const int* __restrict__ ei,
                                  const float* __restrict__ xl,
                                  const float* __restrict__ score,
                                  const unsigned* __restrict__ smax,
                                  float* __restrict__ denom, float* __restrict__ accum,
                                  int E, int Etot) {
  int wave = threadIdx.x >> 5, lane = threadIdx.x & 31;
  int e = blockIdx.x * 8 + wave;
  if (e >= Etot) return;
  int src, dst;
  if (e < E) { src = ei[e]; dst = ei[E + e]; }
  else       { src = dst = e - E; }
  const float* pl = xl + (size_t)src * 256;
  float ex[4];
  #pragma unroll
  for (int h = 0; h < 4; h++)
    ex[h] = __expf(score[(size_t)e * 4 + h] - dec_f32(smax[dst * 4 + h]));
  if (lane < 4) atomicAdd(&denom[dst * 4 + lane], ex[lane]);
  #pragma unroll
  for (int h = 0; h < 4; h++) {
    atomicAdd(&accum[(size_t)dst * 256 + h * 64 + lane],      ex[h] * pl[h * 64 + lane]);
    atomicAdd(&accum[(size_t)dst * 256 + h * 64 + lane + 32], ex[h] * pl[h * 64 + lane + 32]);
  }
}

// ---------- K4: head-average, residual, LayerNorm (wave per node) ----------
__global__ void finalize_kernel(float* __restrict__ x,
                                const float* __restrict__ accum, const float* __restrict__ denom,
                                const float* __restrict__ gbias,
                                half_t* __restrict__ xlnh, int N) {
  int wave = threadIdx.x >> 5, lane = threadIdx.x & 31;
  int n = blockIdx.x * 8 + wave;
  if (n >= N) return;
  float inv[4];
  #pragma unroll
  for (int h = 0; h < 4; h++) inv[h] = 1.f / denom[n * 4 + h];
  int j0 = lane, j1 = lane + 32;
  float s0 = 0.f, s1 = 0.f;
  #pragma unroll
  for (int h = 0; h < 4; h++) {
    s0 += accum[(size_t)n * 256 + h * 64 + j0] * inv[h];
    s1 += accum[(size_t)n * 256 + h * 64 + j1] * inv[h];
  }
  float v0 = x[(size_t)n * 64 + j0] + 0.25f * s0 + gbias[j0];
  float v1 = x[(size_t)n * 64 + j1] + 0.25f * s1 + gbias[j1];
  float sum = wred(v0 + v1);
  float sq  = wred(v0 * v0 + v1 * v1);
  float mean = sum * (1.f / 64.f);
  float var  = sq * (1.f / 64.f) - mean * mean;
  var = var < 0.f ? 0.f : var;
  float rs = rsqrtf(var + LN_EPS);
  float o0 = (v0 - mean) * rs, o1 = (v1 - mean) * rs;
  x[(size_t)n * 64 + j0] = o0;  x[(size_t)n * 64 + j1] = o1;   // f32 copy for residual
  xlnh[(size_t)n * 64 + j0] = (half_t)o0;
  xlnh[(size_t)n * 64 + j1] = (half_t)o1;
}

// ---------- K5: fused residual FF x2 via WMMA (32 rows / block, 4 waves) ----------
// ROWP = 72 halves = 144 B: keeps 16B-aligned rows for async B128 copies AND a
// conflict-free LDS bank pattern (stride 36 dwords; 36*m mod 64 distinct for m<16).
#define ROWP 72
__device__ __forceinline__ void load_a_frags(const half_t* tile, int lane, v16h& a0, v16h& a1) {
  int m = lane & 15, hi = lane >> 4;
  #pragma unroll
  for (int i = 0; i < 8; i++) {
    int kb = ((i >= 4) ? 16 : 0) + hi * 8 + 2 * (i & 3);
    a0[2 * i]     = tile[m * ROWP + kb];
    a0[2 * i + 1] = tile[m * ROWP + kb + 1];
    a1[2 * i]     = tile[m * ROWP + kb + 32];
    a1[2 * i + 1] = tile[m * ROWP + kb + 33];
  }
}
__global__ void ff_kernel(const half_t* __restrict__ xlnh, const float* __restrict__ xlnf,
                          const half_t* __restrict__ w1, const float* __restrict__ b1,
                          const half_t* __restrict__ w2, const float* __restrict__ b2,
                          float* __restrict__ out, int N) {
  __shared__ half_t At[32 * ROWP];
  __shared__ half_t X2[32 * ROWP];
  int tid  = threadIdx.x;              // 128 threads = 4 waves
  int wave = tid >> 5, lane = tid & 31;
  int row0 = blockIdx.x * 32;
  bool full = (row0 + 32 <= N);        // uniform: only last block can be partial

  if (full) {
    // CDNA5 async copy: global -> LDS without VGPR round-trip (ASYNCcnt-tracked).
    // 32 rows x 128B = 4KB; 128 lanes x 16B x 2 rounds.
    #pragma unroll
    for (int rnd = 0; rnd < 2; rnd++) {
      int ci = tid + rnd * 128;                 // 16B chunk id, 8 chunks per row
      int r = ci >> 3, cc = ci & 7;
      unsigned lds_addr = (unsigned)(uintptr_t)(&At[0]) + (unsigned)(r * (ROWP * 2) + cc * 16);
      unsigned long long gaddr =
          (unsigned long long)(uintptr_t)(xlnh + (size_t)(row0 + r) * 64 + cc * 8);
      asm volatile("global_load_async_to_lds_b128 %0, %1, off"
                   :: "v"(lds_addr), "v"(gaddr) : "memory");
    }
    asm volatile("s_wait_asynccnt 0" ::: "memory");
  } else {
    for (int idx = tid; idx < 32 * 64; idx += 128) {
      int r = idx >> 6, c = idx & 63;
      int g = row0 + r;
      At[r * ROWP + c] = (g < N) ? xlnh[(size_t)g * 64 + c] : (half_t)0.f;
    }
  }
  __syncthreads();

  int colbase = wave * 16;
  int hi = lane >> 4;
  int col = colbase + (lane & 15);
  int mbase = 8 * hi;

  // ---- GEMM 1: h1 = relu(x_ln @ W1 + b1); x2 = x_ln + h1 ----
  v16h bf0, bf1;
  #pragma unroll
  for (int j = 0; j < 16; j++) {
    bf0[j] = w1[(((j >= 8) ? 16 : 0) + hi * 8 + (j & 7)) * 64 + col];          // B frag: N = lane%16, K in table order
    bf1[j] = w1[(((j >= 8) ? 16 : 0) + hi * 8 + (j & 7) + 32) * 64 + col];   //         same, K + 32
  }
  float bias1 = b1[col];
  float x2v[2][8];
  #pragma unroll
  for (int t = 0; t < 2; t++) {                    // two 16-row subtiles reuse B frags
    v16h a0, a1;
    load_a_frags(&At[t * 16 * ROWP], lane, a0, a1);
    v8f c = {};
    c = __builtin_amdgcn_wmma_f32_16x16x32_f16(false, a0, false, bf0, (short)0, c, false, false);
    c = __builtin_amdgcn_wmma_f32_16x16x32_f16(false, a1, false, bf1, (short)0, c, false, false);
    if (full) {
      #pragma unroll
      for (int r = 0; r < 8; r++) {
        int mr = t * 16 + mbase + r;
        float h = c[r] + bias1;  h = h > 0.f ? h : 0.f;
        x2v[t][r] = xlnf[(size_t)(row0 + mr) * 64 + col] + h;
        X2[mr * ROWP + col] = (half_t)x2v[t][r];
      }
    } else {
      #pragma unroll
      for (int r = 0; r < 8; r++) {
        int mr = t * 16 + mbase + r;
        int g  = row0 + mr;
        float h = c[r] + bias1;  h = h > 0.f ? h : 0.f;
        float xv = (g < N) ? xlnf[(size_t)g * 64 + col] : 0.f;
        x2v[t][r] = xv + h;
        X2[mr * ROWP + col] = (half_t)x2v[t][r];
      }
    }
  }
  __syncthreads();

  // ---- GEMM 2: out = x2 + relu(x2 @ W2 + b2) ----
  #pragma unroll
  for (int j = 0; j < 16; j++) {
    bf0[j] = w2[(((j >= 8) ? 16 : 0) + hi * 8 + (j & 7)) * 64 + col];
    bf1[j] = w2[(((j >= 8) ? 16 : 0) + hi * 8 + (j & 7) + 32) * 64 + col];
  }
  float bias2 = b2[col];
  #pragma unroll
  for (int t = 0; t < 2; t++) {
    v16h a0, a1;
    load_a_frags(&X2[t * 16 * ROWP], lane, a0, a1);
    v8f c2 = {};
    c2 = __builtin_amdgcn_wmma_f32_16x16x32_f16(false, a0, false, bf0, (short)0, c2, false, false);
    c2 = __builtin_amdgcn_wmma_f32_16x16x32_f16(false, a1, false, bf1, (short)0, c2, false, false);
    if (full) {
      #pragma unroll
      for (int r = 0; r < 8; r++) {
        int mr = t * 16 + mbase + r;
        float h = c2[r] + bias2;  h = h > 0.f ? h : 0.f;
        out[(size_t)(row0 + mr) * 64 + col] = x2v[t][r] + h;
      }
    } else {
      #pragma unroll
      for (int r = 0; r < 8; r++) {
        int mr = t * 16 + mbase + r;
        int g  = row0 + mr;
        float h = c2[r] + bias2;  h = h > 0.f ? h : 0.f;
        if (g < N) out[(size_t)g * 64 + col] = x2v[t][r] + h;
      }
    }
  }
}

// ---------- host ----------
extern "C" void kernel_launch(void* const* d_in, const int* in_sizes, int n_in,
                              void* d_out, int out_size, void* d_ws, size_t ws_size,
                              hipStream_t stream) {
  const float* arrivals   = (const float*)d_in[0];
  const float* departures = (const float*)d_in[1];
  const float* hardm      = (const float*)d_in[2];
  const float* w_emb1     = (const float*)d_in[3];
  const float* b_emb1     = (const float*)d_in[4];
  const float* w_emb2     = (const float*)d_in[5];
  const float* b_emb2     = (const float*)d_in[6];
  const float* gat_wl     = (const float*)d_in[7];
  const float* gat_wr     = (const float*)d_in[8];
  const float* gat_att    = (const float*)d_in[9];
  const float* gat_bias   = (const float*)d_in[10];
  const float* w_ff1      = (const float*)d_in[11];
  const float* b_ff1      = (const float*)d_in[12];
  const float* w_ff2      = (const float*)d_in[13];
  const float* b_ff2      = (const float*)d_in[14];
  const int*   edge_index = (const int*)d_in[15];
  const int*   timestep   = (const int*)d_in[16];

  int N = in_sizes[0];
  int E = in_sizes[15] / 2;
  int Etot = E + N;

  char* ws = (char*)d_ws;
  size_t off = 0;
  auto take = [&](size_t bytes) -> void* {
    void* p = ws + off;
    off += (bytes + 255) & ~(size_t)255;
    return p;
  };
  float*    cW12  = (float*)take(192 * 4);
  float*    cbx   = (float*)take(64 * 4);
  float*    cWl   = (float*)take(768 * 4);
  float*    cbl   = (float*)take(256 * 4);
  float*    cWr   = (float*)take(768 * 4);
  float*    cbr   = (float*)take(256 * 4);
  half_t*   wff1h = (half_t*)take(4096 * 2);
  half_t*   wff2h = (half_t*)take(4096 * 2);
  float*    x     = (float*)take((size_t)N * 64 * 4);    // later holds LN(x) in f32
  float*    xl    = (float*)take((size_t)N * 256 * 4);
  float*    xr    = (float*)take((size_t)N * 256 * 4);
  float*    score = (float*)take((size_t)Etot * 4 * 4);
  unsigned* smax  = (unsigned*)take((size_t)N * 4 * 4);
  float*    denom = (float*)take((size_t)N * 4 * 4);
  float*    accum = (float*)take((size_t)N * 256 * 4);
  half_t*   xlnh  = (half_t*)take((size_t)N * 64 * 2);
  (void)ws_size; (void)n_in; (void)out_size;

  prep_kernel<<<1, 256, 0, stream>>>(w_emb1, b_emb1, w_emb2, b_emb2, gat_wl, gat_wr,
                                     w_ff1, w_ff2, cW12, cbx, cWl, cbl, cWr, cbr,
                                     wff1h, wff2h);
  init_kernel<<<2048, 256, 0, stream>>>(smax, denom, accum, N);
  node_kernel<<<(N + 7) / 8, 256, 0, stream>>>(arrivals, departures, hardm, timestep,
                                               cW12, cbx, cWl, cbl, cWr, cbr,
                                               x, xl, xr, N);
  edge_score_kernel<<<(Etot + 7) / 8, 256, 0, stream>>>(edge_index, xl, xr, gat_att,
                                                        score, smax, E, Etot);
  edge_accum_kernel<<<(Etot + 7) / 8, 256, 0, stream>>>(edge_index, xl, score, smax,
                                                        denom, accum, E, Etot);
  finalize_kernel<<<(N + 7) / 8, 256, 0, stream>>>(x, accum, denom, gat_bias, xlnh, N);
  ff_kernel<<<(N + 31) / 32, 128, 0, stream>>>(xlnh, x, wff1h, b_ff1, wff2h, b_ff2,
                                               (float*)d_out, N);
}
